// RelationalDiscoveryLoss_33655363731899
// MI455X (gfx1250) — matrix operand order repaired and checked by reading, not compiled, not run on hardware
//
#include <hip/hip_runtime.h>
#include <hip/hip_bf16.h>

typedef __attribute__((ext_vector_type(16))) __bf16 v16bf;
typedef __attribute__((ext_vector_type(8)))  __bf16 v8bf;
typedef __attribute__((ext_vector_type(8)))  float  v8f;

#define N_NODES 8192
#define DIM     256
#define B_ROWS  4096
#define N_CLS   1000
#define COS_EPS 1e-8f

// ---------------------------------------------------------------- utilities
__device__ __forceinline__ float wave_sum(float v) {
    #pragma unroll
    for (int o = 16; o > 0; o >>= 1) v += __shfl_xor(v, o, 32);
    return v;
}
__device__ __forceinline__ float wave_max(float v) {
    #pragma unroll
    for (int o = 16; o > 0; o >>= 1) v = fmaxf(v, __shfl_xor(v, o, 32));
    return v;
}

// ---------------------------------------------------------------- kernel 1
__global__ void zero_acc_kernel(float* acc) {
    if (threadIdx.x < 3) acc[threadIdx.x] = 0.0f;
}

// ---------------------------------------------------------------- kernel 2
// Cross-entropy: one block (256 thr = 8 waves) per row of pred.
__global__ __launch_bounds__(256) void ce_kernel(const float* __restrict__ pred,
                                                 const int* __restrict__ target,
                                                 float* __restrict__ acc) {
    const int row  = blockIdx.x;
    const float* p = pred + (size_t)row * N_CLS;
    const int lane = threadIdx.x & 31;
    const int wv   = threadIdx.x >> 5;

    __shared__ float smax[8];
    __shared__ float ssum[8];

    float m = -3.402823466e38f;
    for (int c = threadIdx.x; c < N_CLS; c += 256) m = fmaxf(m, p[c]);
    m = wave_max(m);
    if (lane == 0) smax[wv] = m;
    __syncthreads();
    float bm = smax[0];
    #pragma unroll
    for (int w = 1; w < 8; w++) bm = fmaxf(bm, smax[w]);

    float s = 0.0f;
    for (int c = threadIdx.x; c < N_CLS; c += 256) s += __expf(p[c] - bm);
    s = wave_sum(s);
    if (lane == 0) ssum[wv] = s;
    __syncthreads();
    if (threadIdx.x == 0) {
        float tot = 0.0f;
        #pragma unroll
        for (int w = 0; w < 8; w++) tot += ssum[w];
        float loss = __logf(tot) + bm - p[target[row]];
        atomicAdd(&acc[0], loss);
    }
}

// ---------------------------------------------------------------- kernel 3
// Per-row L2 norm + fp32 -> bf16 conversion. One block (256 thr) per row.
__global__ __launch_bounds__(256) void prep_kernel(const float* __restrict__ E,
                                                   __bf16* __restrict__ Ebf,
                                                   float* __restrict__ norms) {
    const int row = blockIdx.x;
    const int t   = threadIdx.x;
    const size_t idx = (size_t)row * DIM + t;
    float v = E[idx];
    Ebf[idx] = (__bf16)v;                 // RNE conversion
    float sq = wave_sum(v * v);
    __shared__ float ss[8];
    if ((t & 31) == 0) ss[t >> 5] = sq;
    __syncthreads();
    if (t == 0) {
        float tot = 0.0f;
        #pragma unroll
        for (int w = 0; w < 8; w++) tot += ss[w];
        norms[row] = sqrtf(tot);
    }
}

// ---------------------------------------------------------------- kernel 4
// Fused E*E^T (bf16 WMMA) + adj pass. Block: 256 thr = 8 waves.
// Block tile: 128 rows x 64 cols. Wave w: rows ib..ib+15, 4 column tiles.
__global__ __launch_bounds__(256) void
fused_sim_kernel(const __bf16* __restrict__ Ebf,
                 const float* __restrict__ norms,
                 const float* __restrict__ adj,
                 float* __restrict__ acc) {
    const int wv   = threadIdx.x >> 5;
    const int lane = threadIdx.x & 31;
    const int hl   = lane >> 4;          // half-wave: 0 or 1
    const int l16  = lane & 15;

    const int j0 = blockIdx.x * 64;      // gridDim.x = 128
    const int i0 = blockIdx.y * 128;     // gridDim.y = 64
    const int ib = i0 + wv * 16;

    v8f c0 = {}, c1 = {}, c2 = {}, c3 = {};

    // A fragment source: row (ib + l16); k order used here (same for B):
    //   lanes 0-15 hold K=0..15, lanes 16-31 hold K=16..31 (contiguous per lane)
    const __bf16* arow = Ebf + (size_t)(ib + l16) * DIM;
    // B fragment source: column j maps to row j of E; k order used here (same as A):
    //   lanes 0-15 hold K=0..15, lanes 16-31 hold K=16..31 (contiguous per lane)
    const __bf16* brow0 = Ebf + (size_t)(j0 +  0 + l16) * DIM;
    const __bf16* brow1 = Ebf + (size_t)(j0 + 16 + l16) * DIM;
    const __bf16* brow2 = Ebf + (size_t)(j0 + 32 + l16) * DIM;
    const __bf16* brow3 = Ebf + (size_t)(j0 + 48 + l16) * DIM;

    #pragma unroll
    for (int ks = 0; ks < DIM; ks += 32) {
        v8bf alo = *(const v8bf*)(arow + ks + hl * 16);
        v8bf ahi = *(const v8bf*)(arow + ks + hl * 16 + 8);
        v16bf a = __builtin_shufflevector(alo, ahi,
                    0,1,2,3,4,5,6,7,8,9,10,11,12,13,14,15);

        const int kb = ks + hl * 16;
        v8bf b0lo = *(const v8bf*)(brow0 + kb);
        v8bf b0hi = *(const v8bf*)(brow0 + kb + 8);
        v8bf b1lo = *(const v8bf*)(brow1 + kb);
        v8bf b1hi = *(const v8bf*)(brow1 + kb + 8);
        v8bf b2lo = *(const v8bf*)(brow2 + kb);
        v8bf b2hi = *(const v8bf*)(brow2 + kb + 8);
        v8bf b3lo = *(const v8bf*)(brow3 + kb);
        v8bf b3hi = *(const v8bf*)(brow3 + kb + 8);

        v16bf b0 = __builtin_shufflevector(b0lo, b0hi,
                    0,1,2,3,4,5,6,7,8,9,10,11,12,13,14,15);
        v16bf b1 = __builtin_shufflevector(b1lo, b1hi,
                    0,1,2,3,4,5,6,7,8,9,10,11,12,13,14,15);
        v16bf b2 = __builtin_shufflevector(b2lo, b2hi,
                    0,1,2,3,4,5,6,7,8,9,10,11,12,13,14,15);
        v16bf b3 = __builtin_shufflevector(b3lo, b3hi,
                    0,1,2,3,4,5,6,7,8,9,10,11,12,13,14,15);

        c0 = __builtin_amdgcn_wmma_f32_16x16x32_bf16(false, a, false, b0,
                                                     (short)0, c0, false, false);
        c1 = __builtin_amdgcn_wmma_f32_16x16x32_bf16(false, a, false, b1,
                                                     (short)0, c1, false, false);
        c2 = __builtin_amdgcn_wmma_f32_16x16x32_bf16(false, a, false, b2,
                                                     (short)0, c2, false, false);
        c3 = __builtin_amdgcn_wmma_f32_16x16x32_bf16(false, a, false, b3,
                                                     (short)0, c3, false, false);
    }

    // Epilogue: C layout -> (gi, gj); fuse adj read + both reductions.
    const int gjb = j0 + l16;
    const float nj0 = norms[gjb];
    const float nj1 = norms[gjb + 16];
    const float nj2 = norms[gjb + 32];
    const float nj3 = norms[gjb + 48];

    float sem = 0.0f, sab = 0.0f;
    #pragma unroll
    for (int r = 0; r < 8; r++) {
        const int gi = ib + r + 8 * hl;
        const float ni = norms[gi];
        const float* arowj = adj + (size_t)gi * N_NODES + gjb;

        float a0 = arowj[0];
        float a1 = arowj[16];
        float a2 = arowj[32];
        float a3 = arowj[48];

        sem += a0 * (1.0f - c0[r] / fmaxf(ni * nj0, COS_EPS));
        sem += a1 * (1.0f - c1[r] / fmaxf(ni * nj1, COS_EPS));
        sem += a2 * (1.0f - c2[r] / fmaxf(ni * nj2, COS_EPS));
        sem += a3 * (1.0f - c3[r] / fmaxf(ni * nj3, COS_EPS));
        sab += fabsf(a0) + fabsf(a1) + fabsf(a2) + fabsf(a3);
    }

    sem = wave_sum(sem);
    sab = wave_sum(sab);
    __shared__ float s_sem[8];
    __shared__ float s_ab[8];
    if (lane == 0) { s_sem[wv] = sem; s_ab[wv] = sab; }
    __syncthreads();
    if (threadIdx.x == 0) {
        float ts = 0.0f, ta = 0.0f;
        #pragma unroll
        for (int w = 0; w < 8; w++) { ts += s_sem[w]; ta += s_ab[w]; }
        atomicAdd(&acc[1], ta);   // sum |adj|
        atomicAdd(&acc[2], ts);   // sum adj*(1-sim)
    }
}

// ---------------------------------------------------------------- kernel 5
__global__ void finalize_kernel(const float* __restrict__ acc, float* __restrict__ out) {
    const float inv_nn = 1.0f / ((float)N_NODES * (float)N_NODES);
    float l_task     = acc[0] / (float)B_ROWS;
    float l_sparse   = acc[1] * inv_nn;
    float l_semantic = acc[2] * inv_nn;
    out[0] = 1.0f * l_task + 0.01f * l_sparse + 0.1f * l_semantic;
}

// ---------------------------------------------------------------- launch
extern "C" void kernel_launch(void* const* d_in, const int* in_sizes, int n_in,
                              void* d_out, int out_size, void* d_ws, size_t ws_size,
                              hipStream_t stream) {
    (void)in_sizes; (void)n_in; (void)out_size; (void)ws_size;

    const float* pred   = (const float*)d_in[0];
    const int*   target = (const int*)d_in[1];
    const float* adj    = (const float*)d_in[2];
    const float* emb    = (const float*)d_in[3];
    float* out = (float*)d_out;

    char* ws = (char*)d_ws;
    float*  acc   = (float*)ws;                               // 3 floats (pad 256B)
    float*  norms = (float*)(ws + 256);                       // 8192 floats
    __bf16* Ebf   = (__bf16*)(ws + 256 + N_NODES * 4);        // 8192*256 bf16 = 4MB

    zero_acc_kernel<<<1, 32, 0, stream>>>(acc);
    ce_kernel<<<B_ROWS, 256, 0, stream>>>(pred, target, acc);
    prep_kernel<<<N_NODES, 256, 0, stream>>>(emb, Ebf, norms);

    dim3 grid(N_NODES / 64, N_NODES / 128);                   // 128 x 64 blocks
    fused_sim_kernel<<<grid, 256, 0, stream>>>(Ebf, norms, adj, acc);

    finalize_kernel<<<1, 1, 0, stream>>>(acc, out);
}
